// MoleculeDecoder_15238543966497
// MI455X (gfx1250) — compile-verified, matrix operand order checked by reading, not run on hardware
//
#include <hip/hip_runtime.h>
#include <hip/hip_bf16.h>

// ---------------------------------------------------------------------------
// CDNA5 (gfx1250) GraphGPS forward. All large GEMMs use
// v_wmma_f32_16x16x32_f16 (f16 A/B staged via LDS, f32 accumulate).
// GEMM inputs are kept in f16 (produced by the preceding kernel). f16 tile
// staging goes through GLOBAL_LOAD_ASYNC_TO_LDS_B128 (ASYNCcnt) when the
// toolchain exposes the builtin, else a uint4 copy fallback.
// ---------------------------------------------------------------------------

typedef __attribute__((ext_vector_type(16))) _Float16 v16h;
typedef __attribute__((ext_vector_type(8)))  _Float16 v8h;
typedef __attribute__((ext_vector_type(8)))  float    v8f;
typedef __attribute__((ext_vector_type(4)))  int      v4i;

#define GLOBAL_AS __attribute__((address_space(1)))
#define LDS_AS    __attribute__((address_space(3)))

#if defined(__has_builtin)
#if __has_builtin(__builtin_amdgcn_global_load_async_to_lds_b128)
#define HAVE_ASYNC_LDS 1
#endif
#if __has_builtin(__builtin_amdgcn_s_wait_asynccnt)
#define HAVE_WAIT_ASYNC 1
#endif
#endif
#ifndef HAVE_ASYNC_LDS
#define HAVE_ASYNC_LDS 0
#endif
#ifndef HAVE_WAIT_ASYNC
#define HAVE_WAIT_ASYNC 0
#endif

// 16B per-lane global -> LDS copy (async path when available)
__device__ inline void stage16(const _Float16* gsrc, _Float16* lds) {
#if HAVE_ASYNC_LDS
    __builtin_amdgcn_global_load_async_to_lds_b128(
        (GLOBAL_AS v4i*)gsrc, (LDS_AS v4i*)lds, 0, 0);
#else
    *(uint4*)lds = *(const uint4*)gsrc;
#endif
}

__device__ inline void stage_wait() {
#if HAVE_ASYNC_LDS
#if HAVE_WAIT_ASYNC
    __builtin_amdgcn_s_wait_asynccnt(0);
#else
    asm volatile("s_wait_asynccnt 0x0" ::: "memory");
#endif
#endif
}

// --- WMMA helpers -----------------------------------------------------------

__device__ inline v16h frag_ld(const _Float16* p) {
    v8h lo = *(const v8h*)p;
    v8h hi = *(const v8h*)(p + 16);
    return __builtin_shufflevector(lo, hi, 0,1,2,3,4,5,6,7,8,9,10,11,12,13,14,15);
}

// A (16xK) and transposed-B (W row-major [n][k]) share the same per-lane
// pattern on CDNA5: lane = row (mod 16), K halves {hs*8..hs*8+7, +16..+23},
// hs = lane>>4 (CDNA5 ISA "16-bit A-Matrix 16x32" layout).
__device__ inline v16h lds_frag(const _Float16* base, int stride, int lane) {
    int r  = lane & 15;
    int hs = (lane >> 4) << 3;
    return frag_ld(base + r * stride + hs);
}

__device__ inline v8f wmma_f16(v16h a, v16h b, v8f c) {
    return __builtin_amdgcn_wmma_f32_16x16x32_f16(
        false, a, false, b, (short)0, c, false, false);
}

// --- f32 -> f16 conversion --------------------------------------------------

__global__ void cvt_kernel(const float* __restrict__ s, _Float16* __restrict__ d, int n) {
    int i = blockIdx.x * blockDim.x + threadIdx.x;
    if (i < n) d[i] = (_Float16)s[i];
}

// --- Tiled GEMM: Out[R][CO] = epi( X[R][CI] @ W[CO][CI]^T + bias ) ----------
// Block: 256 threads (8 waves), tile 128 rows x 64 cols; wave = 16-row strip.
// EPI: 0 none, 1 ReLU, 2 Out += result. IN16: X is f16. OUT16: write f16.
// R % 128 == 0, CI % 32 == 0, CO % 64 == 0 at every call site.

template<int EPI, int IN16, int OUT16>
__global__ void __launch_bounds__(256) gemm_kernel(
    const void* __restrict__ Xin_, const _Float16* __restrict__ Wp,
    const float* __restrict__ bias, void* __restrict__ Out_,
    int K, int Cols)
{
    __shared__ _Float16 As[128 * 32];   // 8 KB
    __shared__ _Float16 Bs[64 * 32];    // 4 KB

    const int t    = threadIdx.x;
    const int lane = t & 31;
    const int wave = t >> 5;
    const int row0 = blockIdx.x * 128;
    const int col0 = blockIdx.y * 64;

    v8f acc[4] = {};

    for (int kk = 0; kk < K; kk += 32) {
        if (IN16) {
            // A tile 128x32 f16: 512 chunks of 8 halves (16B), 2 per thread
            const _Float16* Xin = (const _Float16*)Xin_;
            #pragma unroll
            for (int i = 0; i < 2; ++i) {
                int idx = t + i * 256;
                int r   = idx >> 2;
                int c8  = (idx & 3) * 8;
                stage16(Xin + (size_t)(row0 + r) * K + kk + c8, &As[r * 32 + c8]);
            }
        } else {
            // A tile 128x32 f32 -> f16: 1024 float4, 4 per thread
            const float* Xin = (const float*)Xin_;
            #pragma unroll
            for (int i = 0; i < 4; ++i) {
                int idx = t + i * 256;
                int r   = idx >> 3;
                int c4  = idx & 7;
                float4 v = *(const float4*)(Xin + (size_t)(row0 + r) * K + kk + c4 * 4);
                int o = r * 32 + c4 * 4;
                As[o+0] = (_Float16)v.x; As[o+1] = (_Float16)v.y;
                As[o+2] = (_Float16)v.z; As[o+3] = (_Float16)v.w;
            }
        }
        {   // B tile: W rows col0..col0+63 (f16), 16B per thread
            int j  = t >> 2;
            int c8 = (t & 3) * 8;
            stage16(Wp + (size_t)(col0 + j) * K + kk + c8, &Bs[j * 32 + c8]);
        }
        stage_wait();
        __syncthreads();
        {
            v16h a = lds_frag(&As[(wave * 16) * 32], 32, lane);
            #pragma unroll
            for (int j = 0; j < 4; ++j) {
                v16h b = lds_frag(&Bs[(j * 16) * 32], 32, lane);
                acc[j] = wmma_f16(a, b, acc[j]);
            }
        }
        __syncthreads();
    }

    // epilogue: C/D layout -> row = base + i + 8*(lane>>4), col = lane&15
    const int lr = lane & 15;
    const int hs = lane >> 4;
    #pragma unroll
    for (int j = 0; j < 4; ++j) {
        int col = col0 + j * 16 + lr;
        float bv = bias ? bias[col] : 0.f;
        #pragma unroll
        for (int i = 0; i < 8; ++i) {
            int row = row0 + wave * 16 + hs * 8 + i;
            float v = acc[j][i] + bv;
            if (EPI == 1) v = fmaxf(v, 0.f);
            size_t o = (size_t)row * Cols + col;
            if (OUT16)          ((_Float16*)Out_)[o] = (_Float16)v;
            else if (EPI == 2)  ((float*)Out_)[o] += v;
            else                ((float*)Out_)[o]  = v;
        }
    }
}

// --- Featurizer first layer: relu(BN_eval(f) @ l1.w^T + l1.b) -> f16 --------

__global__ void feat_l1_kernel(
    const float* __restrict__ f,
    const float* __restrict__ bnb, const float* __restrict__ bng,
    const float* __restrict__ rm,  const float* __restrict__ rv,
    const float* __restrict__ w,   const float* __restrict__ bias,
    _Float16* __restrict__ out, int rows, int d)
{
    int idx = blockIdx.x * blockDim.x + threadIdx.x;
    int row = idx >> 8, c = idx & 255;
    if (row >= rows) return;
    float acc = bias[c];
    for (int k = 0; k < d; ++k) {
        float xn = (f[(size_t)row * d + k] - rm[k]) * rsqrtf(rv[k] + 1e-5f) * bng[k] + bnb[k];
        acc += xn * w[c * d + k];
    }
    out[(size_t)row * 256 + c] = (_Float16)fmaxf(acc, 0.f);
}

// --- LayerNorm over H=256 (one wave per row); ADD: out += LN ----------------

template<int ADD>
__global__ void __launch_bounds__(256) ln_kernel(
    const float* __restrict__ in, const float* __restrict__ g,
    const float* __restrict__ b,  float* __restrict__ out, int rows)
{
    int wave = threadIdx.x >> 5, lane = threadIdx.x & 31;
    int row = blockIdx.x * 8 + wave;
    if (row >= rows) return;
    const float* p = in + (size_t)row * 256;
    float v[8], s = 0.f, s2 = 0.f;
    #pragma unroll
    for (int i = 0; i < 8; ++i) {
        v[i] = p[lane + i * 32]; s += v[i]; s2 += v[i] * v[i];
    }
    #pragma unroll
    for (int off = 16; off > 0; off >>= 1) {
        s += __shfl_xor(s, off, 32); s2 += __shfl_xor(s2, off, 32);
    }
    float mean = s * (1.f / 256.f);
    float var  = s2 * (1.f / 256.f) - mean * mean;
    float inv  = rsqrtf(var + 1e-5f);
    #pragma unroll
    for (int i = 0; i < 8; ++i) {
        int c = lane + i * 32;
        float r = (v[i] - mean) * inv * g[c] + b[c];
        size_t o = (size_t)row * 256 + c;
        if (ADD) out[o] += r; else out[o] = r;
    }
}

// --- BatchNorm(eval): out = BN(a + b) (+ c); optional f16 mirror ------------

__global__ void bn_add_kernel(
    const float* __restrict__ a, const float* __restrict__ b,
    const float* __restrict__ c, float* __restrict__ out,
    _Float16* __restrict__ out16,
    const float* __restrict__ g, const float* __restrict__ beta,
    const float* __restrict__ rm, const float* __restrict__ rv,
    size_t total, int cols)
{
    size_t idx = (size_t)blockIdx.x * blockDim.x + threadIdx.x;
    if (idx >= total) return;
    int col = (int)(idx % cols);
    float x = a[idx] + (b ? b[idx] : 0.f);
    float r = (x - rm[col]) * rsqrtf(rv[col] + 1e-5f) * g[col] + beta[col];
    if (c) r += c[idx];
    out[idx] = r;
    if (out16) out16[idx] = (_Float16)r;
}

// --- GIN input: out16 = f16(x + segment_sum(x[src] -> dst)), block/graph ----
// Thread t owns feature column t -> column-exclusive LDS accumulation,
// no atomics / no barriers needed.

__global__ void __launch_bounds__(256) gin_input_kernel(
    const float* __restrict__ x, const int* __restrict__ src,
    const int* __restrict__ dst, _Float16* __restrict__ out)
{
    __shared__ float acc[64 * 256];   // 64 KB
    int g = blockIdx.x, t = threadIdx.x;
    for (int n = 0; n < 64; ++n) acc[n * 256 + t] = 0.f;
    int e0 = g * 256;                 // 256 edges per graph, contiguous
    for (int e = 0; e < 256; ++e) {
        int s = src[e0 + e];
        int d = dst[e0 + e] - g * 64;
        acc[d * 256 + t] += x[(size_t)s * 256 + t];
    }
    for (int n = 0; n < 64; ++n) {
        int node = g * 64 + n;
        out[(size_t)node * 256 + t] =
            (_Float16)(x[(size_t)node * 256 + t] + acc[n * 256 + t]);
    }
}

// --- Attention: block per (graph, head); 4 waves, 16-query strip per wave ---

__global__ void __launch_bounds__(128) attn_kernel(
    const _Float16* __restrict__ qkv, _Float16* __restrict__ Out)
{
    __shared__ _Float16 qs[64 * 64];   // Q (later reused as P)   8 KB
    __shared__ _Float16 ks[64 * 64];   // K row-major [n][d]      8 KB
    __shared__ _Float16 vts[64 * 64];  // V transposed [d][n]     8 KB
    __shared__ float    ss[64 * 64];   // scores                 16 KB

    int g = blockIdx.x >> 2, h = blockIdx.x & 3;
    int t = threadIdx.x, lane = t & 31, wave = t >> 5;

    const _Float16* base = qkv + (size_t)(g * 64) * 768 + h * 64;
    for (int i = t; i < 64 * 64; i += 128) {
        int r = i >> 6, d = i & 63;
        const _Float16* p = base + (size_t)r * 768;
        qs[i]           = p[d];
        ks[i]           = p[256 + d];
        vts[d * 64 + r] = p[512 + d];
    }
    __syncthreads();

    const int r0 = wave * 16;
    const int lr = lane & 15, hs = lane >> 4;

    // S = Q K^T
    v8f s0 = {}, s1 = {}, s2 = {}, s3 = {};
    #pragma unroll
    for (int kk = 0; kk < 64; kk += 32) {
        v16h a  = lds_frag(&qs[r0 * 64 + kk], 64, lane);
        v16h b0 = lds_frag(&ks[ 0 * 64 + kk], 64, lane);
        v16h b1 = lds_frag(&ks[16 * 64 + kk], 64, lane);
        v16h b2 = lds_frag(&ks[32 * 64 + kk], 64, lane);
        v16h b3 = lds_frag(&ks[48 * 64 + kk], 64, lane);
        s0 = wmma_f16(a, b0, s0); s1 = wmma_f16(a, b1, s1);
        s2 = wmma_f16(a, b2, s2); s3 = wmma_f16(a, b3, s3);
    }
    const float scale = 0.125f;   // 1/sqrt(dh=64)
    #pragma unroll
    for (int i = 0; i < 8; ++i) {
        int row = r0 + hs * 8 + i;
        ss[row * 64 +  0 + lr] = s0[i] * scale;
        ss[row * 64 + 16 + lr] = s1[i] * scale;
        ss[row * 64 + 32 + lr] = s2[i] * scale;
        ss[row * 64 + 48 + lr] = s3[i] * scale;
    }
    __syncthreads();

    // softmax over cols: 2 lanes per row, P -> qs (f16)
    {
        int row = r0 + lr;
        int cb  = hs * 32;
        float* srow = &ss[row * 64 + cb];
        float m = -1e30f;
        for (int c = 0; c < 32; ++c) m = fmaxf(m, srow[c]);
        m = fmaxf(m, __shfl_xor(m, 16, 32));
        float sum = 0.f;
        for (int c = 0; c < 32; ++c) { float e = __expf(srow[c] - m); srow[c] = e; sum += e; }
        sum += __shfl_xor(sum, 16, 32);
        float inv = 1.f / sum;
        for (int c = 0; c < 32; ++c) qs[row * 64 + cb + c] = (_Float16)(srow[c] * inv);
    }
    __syncthreads();

    // O = P V   (vts is [d][n] -> B-transposed pattern again)
    v8f o0 = {}, o1 = {}, o2 = {}, o3 = {};
    #pragma unroll
    for (int nn = 0; nn < 64; nn += 32) {
        v16h a  = lds_frag(&qs[r0 * 64 + nn], 64, lane);
        v16h b0 = lds_frag(&vts[ 0 * 64 + nn], 64, lane);
        v16h b1 = lds_frag(&vts[16 * 64 + nn], 64, lane);
        v16h b2 = lds_frag(&vts[32 * 64 + nn], 64, lane);
        v16h b3 = lds_frag(&vts[48 * 64 + nn], 64, lane);
        o0 = wmma_f16(a, b0, o0); o1 = wmma_f16(a, b1, o1);
        o2 = wmma_f16(a, b2, o2); o3 = wmma_f16(a, b3, o3);
    }
    _Float16* ob = Out + (size_t)(g * 64) * 256 + h * 64;
    #pragma unroll
    for (int i = 0; i < 8; ++i) {
        int row = r0 + hs * 8 + i;
        ob[(size_t)row * 256 +  0 + lr] = (_Float16)o0[i];
        ob[(size_t)row * 256 + 16 + lr] = (_Float16)o1[i];
        ob[(size_t)row * 256 + 32 + lr] = (_Float16)o2[i];
        ob[(size_t)row * 256 + 48 + lr] = (_Float16)o3[i];
    }
}

// --- Edge endpoint gather: out16 = f16(x[src] + x[dst]) ---------------------

__global__ void edge_gather_kernel(
    const float* __restrict__ x, const int* __restrict__ src,
    const int* __restrict__ dst, _Float16* __restrict__ out, size_t total)
{
    size_t idx = (size_t)blockIdx.x * blockDim.x + threadIdx.x;
    if (idx >= total) return;
    size_t e = idx >> 8; int c = (int)(idx & 255);
    out[idx] = (_Float16)(x[(size_t)src[e] * 256 + c] + x[(size_t)dst[e] * 256 + c]);
}

// --- Classifier head + log_softmax (small CO, VALU) -------------------------

template<int C>
__global__ void head_kernel(
    const float* __restrict__ x, const float* __restrict__ w,
    const float* __restrict__ b, float* __restrict__ out, int rows)
{
    int row = blockIdx.x * blockDim.x + threadIdx.x;
    if (row >= rows) return;
    float acc[C];
    #pragma unroll
    for (int c = 0; c < C; ++c) acc[c] = b[c];
    const float* xr = x + (size_t)row * 256;
    for (int k = 0; k < 256; ++k) {
        float xv = xr[k];
        #pragma unroll
        for (int c = 0; c < C; ++c) acc[c] += xv * w[c * 256 + k];
    }
    float m = acc[0];
    #pragma unroll
    for (int c = 1; c < C; ++c) m = fmaxf(m, acc[c]);
    float s = 0.f;
    #pragma unroll
    for (int c = 0; c < C; ++c) s += __expf(acc[c] - m);
    float lse = m + __logf(s);
    #pragma unroll
    for (int c = 0; c < C; ++c) out[(size_t)row * C + c] = acc[c] - lse;
}

// ---------------------------------------------------------------------------

extern "C" void kernel_launch(void* const* d_in, const int* in_sizes, int n_in,
                              void* d_out, int out_size, void* d_ws, size_t ws_size,
                              hipStream_t stream) {
    (void)in_sizes; (void)n_in; (void)out_size; (void)ws_size;
    constexpr int N = 65536, H = 256, G = 1024, E = 262144;

    const float* lap  = (const float*)d_in[0];
    const float* rwse = (const float*)d_in[1];
    const float* lat  = (const float*)d_in[2];

    // params pytree flattened in JAX order (nested dicts sorted by key):
    // edge_head{b,w}, edge_l1{b,w}, edge_l2{b,w},
    // feat{laplace,rwse}x{bn{b,g,rm,rv}, l1{b,w}, l2{b,w}, ln_b, ln_g},
    // layers[4]x{attn_in{b,w}, attn_out{b,w}, gin_l1{b,w}, gin_l2{b,w},
    //            mlp_l1{b,w}, mlp_l2{b,w}, norm1..3{b,g,rm,rv}},
    // node_head{b,w}, noise{b,w}; then edge_index, batch.
    auto F = [&](int i) -> const float* { return (const float*)d_in[i]; };
    const int PB = 3;
    const int EHEAD_B = PB + 0, EHEAD_W = PB + 1;
    const int EL1_B = PB + 2, EL1_W = PB + 3, EL2_B = PB + 4, EL2_W = PB + 5;
    const int FEAT0 = PB + 6;    // bn_b,bn_g,bn_rm,bn_rv,l1_b,l1_w,l2_b,l2_w,ln_b,ln_g
    const int FEAT1 = PB + 16;
    const int LY0 = PB + 26;     // 24 leaves per layer
    const int NODE_B = LY0 + 96, NODE_W = LY0 + 97;
    const int NOI_B  = LY0 + 98, NOI_W  = LY0 + 99;
    const int* srcI = (const int*)d_in[LY0 + 100];
    const int* dstI = srcI + E;

    // ---- workspace layout ----
    const size_t NH = (size_t)N * H;
    float* X = (float*)d_ws;                 // N x 256 f32
    float* A = X + NH;                       // N x 256 f32
    float* B = A + NH;                       // N x 256 f32
    float* C = B + NH;                       // N x 256 f32
    _Float16* H16a = (_Float16*)(C + NH);    // N x 768 f16 (qkv / hidden)
    _Float16* H16b = H16a + (size_t)N * 768; // N x 256 f16 (gemm-input staging)
    _Float16* X16  = H16b + NH;              // f16 mirror of X
    _Float16* C16  = X16 + NH;               // f16 mirror of C
    _Float16* W16  = C16 + NH;               // weight cache
    _Float16* wf0  = W16;
    _Float16* wf1  = wf0 + H * H;
    _Float16* wno  = wf1 + H * H;
    _Float16* wl   = wno + H * 64;           // 4 layers x 655360 halves
    _Float16* we1  = wl + (size_t)4 * 655360;
    _Float16* we2  = we1 + H * H;

    auto cvt = [&](const float* s, _Float16* d, int n) {
        cvt_kernel<<<(n + 255) / 256, 256, 0, stream>>>(s, d, n);
    };
    cvt(F(FEAT0 + 7), wf0, H * H);
    cvt(F(FEAT1 + 7), wf1, H * H);
    cvt(F(NOI_W), wno, H * 64);
    for (int l = 0; l < 4; ++l) {
        _Float16* lw = wl + (size_t)l * 655360;
        int pb = LY0 + l * 24;
        cvt(F(pb + 1),  lw + 0,      768 * H);  // attn_in.w
        cvt(F(pb + 3),  lw + 196608, H * H);    // attn_out.w
        cvt(F(pb + 5),  lw + 262144, H * H);    // gin_l1.w
        cvt(F(pb + 7),  lw + 327680, H * H);    // gin_l2.w
        cvt(F(pb + 9),  lw + 393216, 512 * H);  // mlp_l1.w
        cvt(F(pb + 11), lw + 524288, H * 512);  // mlp_l2.w
    }
    cvt(F(EL1_W), we1, H * H);
    cvt(F(EL2_W), we2, H * H);

    // gemm dispatch: combos <EPI,IN16,OUT16> actually used
    auto gemm = [&](const void* in, const _Float16* w, const float* bias, void* out,
                    int rows, int K, int cols, int epi, int in16, int out16) {
        dim3 grid(rows / 128, cols / 64);
        if (in16 && out16 && epi == 1)
            gemm_kernel<1,1,1><<<grid, 256, 0, stream>>>(in, w, bias, out, K, cols);
        else if (in16 && out16)
            gemm_kernel<0,1,1><<<grid, 256, 0, stream>>>(in, w, bias, out, K, cols);
        else if (in16)
            gemm_kernel<0,1,0><<<grid, 256, 0, stream>>>(in, w, bias, out, K, cols);
        else if (epi == 2)
            gemm_kernel<2,0,0><<<grid, 256, 0, stream>>>(in, w, bias, out, K, cols);
        else
            gemm_kernel<0,0,0><<<grid, 256, 0, stream>>>(in, w, bias, out, K, cols);
    };

    // ---- featurizers: x = sum LN(mlp(BN(f))) + latents @ noise^T + b ----
    feat_l1_kernel<<<N * H / 256, 256, 0, stream>>>(
        lap, F(FEAT0+0), F(FEAT0+1), F(FEAT0+2), F(FEAT0+3), F(FEAT0+5), F(FEAT0+4), H16a, N, 8);
    gemm(H16a, wf0, F(FEAT0 + 6), B, N, H, H, 0, 1, 0);
    ln_kernel<0><<<N / 8, 256, 0, stream>>>(B, F(FEAT0 + 9), F(FEAT0 + 8), X, N);
    feat_l1_kernel<<<N * H / 256, 256, 0, stream>>>(
        rwse, F(FEAT1+0), F(FEAT1+1), F(FEAT1+2), F(FEAT1+3), F(FEAT1+5), F(FEAT1+4), H16a, N, 20);
    gemm(H16a, wf1, F(FEAT1 + 6), B, N, H, H, 0, 1, 0);
    ln_kernel<1><<<N / 8, 256, 0, stream>>>(B, F(FEAT1 + 9), F(FEAT1 + 8), X, N);
    gemm(lat, wno, F(NOI_B), X, N, 64, H, 2, 0, 0);     // X += latents @ noise^T + b
    cvt_kernel<<<(int)(NH / 256), 256, 0, stream>>>(X, X16, (int)NH);

    // ---- GPS layers ----
    for (int l = 0; l < 4; ++l) {
        int pb = LY0 + l * 24;
        _Float16* lw = wl + (size_t)l * 655360;
        const float *n1b=F(pb+12), *n1g=F(pb+13), *n1rm=F(pb+14), *n1rv=F(pb+15);
        const float *n2b=F(pb+16), *n2g=F(pb+17), *n2rm=F(pb+18), *n2rv=F(pb+19);
        const float *n3b=F(pb+20), *n3g=F(pb+21), *n3rm=F(pb+22), *n3rv=F(pb+23);

        // local GIN branch: h_loc = BN1(mlp(x + neigh) + x)   -> B (f32)
        gin_input_kernel<<<G, 256, 0, stream>>>(X, srcI, dstI, H16b);
        gemm(H16b, lw + 262144, F(pb + 4), H16a, N, H, H, 1, 1, 1);
        gemm(H16a, lw + 327680, F(pb + 6), B, N, H, H, 0, 1, 0);
        bn_add_kernel<<<(unsigned)(NH / 256), 256, 0, stream>>>(
            B, X, nullptr, B, nullptr, n1g, n1b, n1rm, n1rv, NH, H);

        // global branch: h_att = BN2(attn_out(attn(qkv(x))) + x); C = h_loc + h_att
        gemm(X16, lw + 0, F(pb + 0), H16a, N, H, 768, 0, 1, 1);   // qkv (f16)
        attn_kernel<<<G * 4, 128, 0, stream>>>(H16a, H16b);
        gemm(H16b, lw + 196608, F(pb + 2), A, N, H, H, 0, 1, 0);  // attn_out
        bn_add_kernel<<<(unsigned)(NH / 256), 256, 0, stream>>>(
            A, X, B, C, C16, n2g, n2b, n2rm, n2rv, NH, H);

        // feedforward: x = BN3(mlp(out) + out)
        gemm(C16, lw + 393216, F(pb + 8), H16a, N, H, 512, 1, 1, 1);
        gemm(H16a, lw + 524288, F(pb + 10), B, N, 512, H, 0, 1, 0);
        bn_add_kernel<<<(unsigned)(NH / 256), 256, 0, stream>>>(
            B, C, nullptr, X, X16, n3g, n3b, n3rm, n3rv, NH, H);
    }

    // ---- heads ----
    float* nodeOut = (float*)d_out;
    float* edgeOut = nodeOut + (size_t)N * 12;
    head_kernel<12><<<N / 256, 256, 0, stream>>>(X, F(NODE_W), F(NODE_B), nodeOut, N);

    for (int c = 0; c < 4; ++c) {            // 4 chunks of 65536 edges
        const int CE = 65536;
        const int* s  = srcI + c * CE;
        const int* d2 = dstI + c * CE;
        edge_gather_kernel<<<(unsigned)((size_t)CE * H / 256), 256, 0, stream>>>(
            X, s, d2, H16b, (size_t)CE * H);
        gemm(H16b, we1, F(EL1_B), H16a, CE, H, H, 1, 1, 1);
        gemm(H16a, we2, F(EL2_B), B, CE, H, H, 0, 1, 0);
        head_kernel<5><<<CE / 256, 256, 0, stream>>>(
            B, F(EHEAD_W), F(EHEAD_B), edgeOut + (size_t)c * CE * 5, CE);
    }
}
